// BatchMultiHeadGraphAttention_12618613916144
// MI455X (gfx1250) — matrix operand order repaired and checked by reading, not compiled, not run on hardware
//
#include <hip/hip_runtime.h>
#include <hip/hip_bf16.h>
#include <math.h>

// Problem constants (from reference)
#define N_HEADS 4
#define F_IN    768
#define F_OUT   64
#define BSZ     4
#define NN      2048

typedef float  v2f   __attribute__((ext_vector_type(2)));
typedef float  v8f   __attribute__((ext_vector_type(8)));
typedef __bf16 v16bf __attribute__((ext_vector_type(16)));

// ---------------------------------------------------------------------------
// Kernel 1: h_prime = h @ w  (fp32 WMMA 16x16x4), then
//   - store transposed bf16 copy of h_prime:  vt[bh][o][n]   (B-fragment friendly)
//   - attn_src[bh][n] = tanh(h_prime) . a_src ; attn_dst likewise
// grid = (16 row-tiles of 128, 16 bh), block = 256 (8 waves, 16 rows/wave)
// ---------------------------------------------------------------------------
__global__ __launch_bounds__(256) void gat_hprime_kernel(
    const float* __restrict__ h, const float* __restrict__ w,
    const float* __restrict__ a_src, const float* __restrict__ a_dst,
    __hip_bfloat16* __restrict__ vt_raw,
    float* __restrict__ asrc_out, float* __restrict__ adst_out)
{
    __bf16* vt = reinterpret_cast<__bf16*>(vt_raw);
    const int bh   = blockIdx.y;            // b*N_HEADS + head
    const int b    = bh >> 2;
    const int head = bh & 3;
    const int wave = threadIdx.x >> 5;
    const int lane = threadIdx.x & 31;
    const int hi   = lane >> 4;             // lane half (K split)
    const int lm   = lane & 15;             // row (A) / col (B,C,D) index
    const int row0 = blockIdx.x * 128 + wave * 16;

    const float* hrow = h + ((size_t)b * NN + row0 + lm) * F_IN;
    const float* wp   = w + (size_t)head * F_IN * F_OUT;

    v8f acc[4] = {};                        // 16x64 fp32 accumulator (4 N-tiles)

    for (int k = 0; k < F_IN; k += 4) {
        // A fragment (16x4 f32): lane lm holds row, VGPR0/1 = K = k+2*hi, k+2*hi+1
        v2f a = *(const v2f*)(hrow + k + 2 * hi);
        __builtin_prefetch(hrow + k + 64, 0, 1);   // global_prefetch_b8
#pragma unroll
        for (int nt = 0; nt < 4; ++nt) {
            const int n = nt * 16 + lm;
            v2f bb;                         // B fragment (4x16 f32)
            bb.x = wp[(k + 2 * hi    ) * F_OUT + n];
            bb.y = wp[(k + 2 * hi + 1) * F_OUT + n];
            acc[nt] = __builtin_amdgcn_wmma_f32_16x16x4_f32(
                false, a, false, bb, (short)0, acc[nt], false, false);
        }
    }

    // Per-lane slice of a_src / a_dst for this lane's columns
    float asv[4], adv[4];
#pragma unroll
    for (int nt = 0; nt < 4; ++nt) {
        asv[nt] = a_src[head * F_OUT + nt * 16 + lm];
        adv[nt] = a_dst[head * F_OUT + nt * 16 + lm];
    }

#pragma unroll
    for (int r = 0; r < 8; ++r) {
        const int row = row0 + r + 8 * hi;  // C/D layout: VGPR r -> M = r (+8 for hi half)
        float ps = 0.f, pd = 0.f;
#pragma unroll
        for (int nt = 0; nt < 4; ++nt) {
            const float hp  = acc[nt][r];
            const int   col = nt * 16 + lm;
            vt[((size_t)bh * F_OUT + col) * NN + row] = (__bf16)hp;  // transposed bf16 V
            const float t = __builtin_amdgcn_tanhf(hp);              // native v_tanh_f32
            ps += t * asv[nt];
            pd += t * adv[nt];
        }
        // reduce across the 16 lanes of this half-wave (columns)
#pragma unroll
        for (int m = 8; m >= 1; m >>= 1) {
            ps += __shfl_xor(ps, m, 32);
            pd += __shfl_xor(pd, m, 32);
        }
        if (lm == 0) {
            asrc_out[bh * NN + row] = ps;
            adst_out[bh * NN + row] = pd;
        }
    }
}

// ---------------------------------------------------------------------------
// Kernel 2: flash-style attention without materializing the 2048x2048 matrix.
//   Row max is exact & free: max_j leaky(s_i + d_j) = leaky(s_i + max_j d_j).
//   Pass A: per-row softmax denominator l_i.
//   Pass B: P tiles (16x32 bf16, exp of logits built in registers) x
//           V tiles (32x16 bf16, single 256-bit vector load each) via WMMA,
//           fp32 accumulate, scale by 1/l_i, add bias.
// grid = (16 row-tiles of 128, 16 bh), block = 256 (8 waves, 16 rows/wave)
// ---------------------------------------------------------------------------
__global__ __launch_bounds__(256) void gat_attn_kernel(
    const __hip_bfloat16* __restrict__ vt_raw,
    const float* __restrict__ asrc, const float* __restrict__ adst,
    const float* __restrict__ bias, float* __restrict__ out)
{
    const __bf16* vt = reinterpret_cast<const __bf16*>(vt_raw);
    __shared__ float rcp_l_sh[8 * 16];

    const int bh   = blockIdx.y;
    const int wave = threadIdx.x >> 5;
    const int lane = threadIdx.x & 31;
    const int hi   = lane >> 4;
    const int lm   = lane & 15;
    const int row0 = blockIdx.x * 128 + wave * 16;

    const float* dptr = adst + bh * NN;
    const float* sptr = asrc + bh * NN;
    const float  L2E  = 1.44269504088896f;

    // global max of dst column term (shared by every row of this (b,h))
    float dm = -1e30f;
    for (int j = lane; j < NN; j += 32) dm = fmaxf(dm, dptr[j]);
#pragma unroll
    for (int m = 16; m >= 1; m >>= 1) dm = fmaxf(dm, __shfl_xor(dm, m, 32));

    // Pass A: softmax denominators for this wave's 16 rows
    float my_s = 0.f, my_m = 0.f;
    for (int r = 0; r < 16; ++r) {
        const float s  = sptr[row0 + r];
        float xm = s + dm; xm = (xm >= 0.f) ? xm : 0.2f * xm;   // exact row max
        float l = 0.f;
        for (int j = lane; j < NN; j += 32) {
            float x = s + dptr[j];
            x = (x >= 0.f) ? x : 0.2f * x;
            l += __builtin_amdgcn_exp2f((x - xm) * L2E);
        }
#pragma unroll
        for (int m = 16; m >= 1; m >>= 1) l += __shfl_xor(l, m, 32);
        if (lm == r) { my_s = s; my_m = xm; }    // both halves keep row lm stats
        if (lane == r) rcp_l_sh[wave * 16 + r] = __builtin_amdgcn_rcpf(l);
    }
    __syncthreads();

    // Pass B: P @ V with bf16 WMMA, fp32 accumulate
    v8f acc[4] = {};
    const __bf16* vbase = vt + (size_t)bh * F_OUT * NN;

    for (int j0 = 0; j0 < NN; j0 += 32) {
        // A fragment: P tile 16x32 bf16; lane lm = row, K = 16*hi + s (matches B)
        v16bf pa;
#pragma unroll
        for (int v = 0; v < 8; ++v) {
            const int kb = 16 * hi + 2 * v;
            v2f d = *(const v2f*)(dptr + j0 + kb);          // contiguous pair
            float x0 = my_s + d.x;
            float x1 = my_s + d.y;
            x0 = (x0 >= 0.f) ? x0 : 0.2f * x0;
            x1 = (x1 >= 0.f) ? x1 : 0.2f * x1;
            pa[2 * v    ] = (__bf16)__builtin_amdgcn_exp2f((x0 - my_m) * L2E);
            pa[2 * v + 1] = (__bf16)__builtin_amdgcn_exp2f((x1 - my_m) * L2E);
        }
#pragma unroll
        for (int nt = 0; nt < 4; ++nt) {
            const int n = nt * 16 + lm;
            // B fragment: V tile 32x16 bf16. Lane's 16 elements are K=16*hi..16*hi+15,
            // which is exactly memory order in vt[o][j] and 32B-aligned: one vector load.
            const v16bf bB = *(const v16bf*)(vbase + (size_t)n * NN + j0 + 16 * hi);
            acc[nt] = __builtin_amdgcn_wmma_f32_16x16x32_bf16(
                false, pa, false, bB, (short)0, acc[nt], false, false);
        }
    }

    // normalize, add bias, store
#pragma unroll
    for (int r = 0; r < 8; ++r) {
        const int   row = row0 + r + 8 * hi;
        const float rl  = rcp_l_sh[wave * 16 + r + 8 * hi];
#pragma unroll
        for (int nt = 0; nt < 4; ++nt) {
            const int col = nt * 16 + lm;
            out[((size_t)bh * NN + row) * F_OUT + col] = acc[nt][r] * rl + bias[col];
        }
    }
}

// ---------------------------------------------------------------------------
extern "C" void kernel_launch(void* const* d_in, const int* in_sizes, int n_in,
                              void* d_out, int out_size, void* d_ws, size_t ws_size,
                              hipStream_t stream) {
    (void)in_sizes; (void)n_in; (void)out_size; (void)ws_size;
    const float* h     = (const float*)d_in[0];
    const float* w     = (const float*)d_in[1];
    const float* a_src = (const float*)d_in[2];
    const float* a_dst = (const float*)d_in[3];
    const float* bias  = (const float*)d_in[4];

    char* ws = (char*)d_ws;
    __hip_bfloat16* vt = (__hip_bfloat16*)ws;                 // 16*64*2048*2B = 4 MiB
    float* asrc_out    = (float*)(ws + (size_t)16 * F_OUT * NN * 2);
    float* adst_out    = asrc_out + 16 * NN;                  // 128 KiB each
    float* outp        = (float*)d_out;

    dim3 grid(16, 16), block(256);
    gat_hprime_kernel<<<grid, block, 0, stream>>>(h, w, a_src, a_dst, vt, asrc_out, adst_out);
    gat_attn_kernel  <<<grid, block, 0, stream>>>(vt, asrc_out, adst_out, bias, outp);
}
